// RTDecode_19353122636501
// MI455X (gfx1250) — matrix operand order repaired and checked by reading, not compiled, not run on hardware
//
#include <hip/hip_runtime.h>
#include <math.h>

typedef _Float16 v16h __attribute__((ext_vector_type(16)));
typedef _Float16 v8h  __attribute__((ext_vector_type(8)));
typedef float    v8f  __attribute__((ext_vector_type(8)));
typedef int      v4i  __attribute__((ext_vector_type(4)));

#define HD   512          // hidden size
#define BAT  128          // batch
#define TT   512          // time steps
#define K2   1024         // 2*HD (fused [x|h] K dim)
#define G4   2048         // 4*HD (gates)
#define MIDN 256          // head mid
#define NROWS (BAT*TT)    // 65536

// ---- CDNA5 async global->LDS staging (guarded; sync fallback) -------------
#if defined(__has_builtin)
#  if __has_builtin(__builtin_amdgcn_global_load_async_to_lds_b128)
#    define HAVE_ASYNC_LDS 1
#  endif
#endif

static __device__ __forceinline__ void async_stage_b128(const _Float16* src,
                                                        _Float16* dst_lds) {
#if defined(HAVE_ASYNC_LDS)
  __builtin_amdgcn_global_load_async_to_lds_b128(
      (__attribute__((address_space(1))) v4i*)src,
      (__attribute__((address_space(3))) v4i*)dst_lds, 0, 0);
#else
  *(v8h*)dst_lds = *(const v8h*)src;   // sync: global load + ds store
#endif
}

static __device__ __forceinline__ void wait_async0() {
#if defined(HAVE_ASYNC_LDS)
#  if __has_builtin(__builtin_amdgcn_s_wait_asynccnt)
  __builtin_amdgcn_s_wait_asynccnt(0);
#  else
  asm volatile("s_wait_asynccnt 0x0" ::: "memory");
#  endif
#endif
}

static __device__ __forceinline__ v16h cat8(v8h lo, v8h hi) {
  return __builtin_shufflevector(lo, hi, 0,1,2,3,4,5,6,7,8,9,10,11,12,13,14,15);
}
static __device__ __forceinline__ float sigmf(float x) { return 1.0f / (1.0f + expf(-x)); }

// ---------------------------------------------------------------------------
// Weight prep: transpose + fp32->fp16.
// Wt{0,1}[k, n] (k<512: W_ih.T row k; k>=512: W_hh.T row k-512), n over 4H.
// Wh1[k, n] = W1.T  ([512,256])
// ---------------------------------------------------------------------------
__global__ __launch_bounds__(256) void prep_weights(
    const float* __restrict__ Wih0, const float* __restrict__ Whh0,
    const float* __restrict__ Wih1, const float* __restrict__ Whh1,
    const float* __restrict__ W1,
    _Float16* __restrict__ Wt0, _Float16* __restrict__ Wt1,
    _Float16* __restrict__ Wh1)
{
  const int NT0 = K2 * G4;          // 2097152
  const int NH  = HD * MIDN;        // 131072
  const int total = 2 * NT0 + NH;
  for (int idx = blockIdx.x * blockDim.x + threadIdx.x; idx < total;
       idx += gridDim.x * blockDim.x) {
    if (idx < NT0) {
      int k = idx >> 11, nn = idx & 2047;
      float v = (k < HD) ? Wih0[(size_t)nn * HD + k] : Whh0[(size_t)nn * HD + (k - HD)];
      Wt0[idx] = (_Float16)v;
    } else if (idx < 2 * NT0) {
      int i = idx - NT0;
      int k = i >> 11, nn = i & 2047;
      float v = (k < HD) ? Wih1[(size_t)nn * HD + k] : Whh1[(size_t)nn * HD + (k - HD)];
      Wt1[i] = (_Float16)v;
    } else {
      int i = idx - 2 * NT0;
      int k = i >> 8, nn = i & 255;
      Wh1[i] = (_Float16)W1[(size_t)nn * HD + k];
    }
  }
}

// ---------------------------------------------------------------------------
// State init: xh0 = [rt | 0], xh1 = 0, c = 0, fused biases, zero BN sums.
// ---------------------------------------------------------------------------
__global__ __launch_bounds__(256) void init_state(
    const float* __restrict__ rt,
    const float* __restrict__ bih0, const float* __restrict__ bhh0,
    const float* __restrict__ bih1, const float* __restrict__ bhh1,
    _Float16* __restrict__ xh0, _Float16* __restrict__ xh1,
    float* __restrict__ c0, float* __restrict__ c1,
    float* __restrict__ bias0, float* __restrict__ bias1,
    float* __restrict__ sums, float* __restrict__ sumsq)
{
  int i = blockIdx.x * blockDim.x + threadIdx.x;
  if (i < BAT * K2) {
    int m = i >> 10, k = i & 1023;
    xh0[i] = (k < HD) ? (_Float16)rt[m * HD + k] : (_Float16)0.0f;
    xh1[i] = (_Float16)0.0f;
  }
  if (i < BAT * HD) { c0[i] = 0.0f; c1[i] = 0.0f; }
  if (i < G4)   { bias0[i] = bih0[i] + bhh0[i]; bias1[i] = bih1[i] + bhh1[i]; }
  if (i < MIDN) { sums[i] = 0.0f; sumsq[i] = 0.0f; }
}

// ---------------------------------------------------------------------------
// One LSTM layer step: gates = [x|h] @ Wt + bias, fused cell update.
// grid = (32 n-strips of 16 hidden cols, 2 m-groups of 64 rows), block = 128.
// Each wave owns one 16-row M tile and all 4 gates of a 16-col strip
// (4 WMMA accumulators), so the nonlinearity fuses into the epilogue.
// B tiles double-buffered in LDS; next chunk staged (async if available)
// while WMMAs consume the current one -> one barrier per k-iteration.
// ---------------------------------------------------------------------------
__global__ __launch_bounds__(128) void lstm_step(
    const _Float16* __restrict__ xh,     // [128,1024] fp16 [x|h]
    const _Float16* __restrict__ Wt,     // [1024,2048] fp16
    const float*    __restrict__ bias,   // [2048]
    float*          __restrict__ Cst,    // [128,512] cell state (in/out)
    _Float16*       __restrict__ hdstA,  // h -> next consumer x region (stride 1024)
    _Float16*       __restrict__ hdstB,  // h -> own h region (base already +512)
    _Float16*       __restrict__ feats)  // null, or feats + t*512 (row stride T*HD)
{
  __shared__ __align__(16) _Float16 Bsh[2][4 * 32 * 16];   // 2 x 4KB [gate][k][n]
  const int lane  = threadIdx.x & 31;
  const int w     = threadIdx.x >> 5;
  const int nbase = blockIdx.x * 16;
  const int mtile = blockIdx.y * 4 + w;

  v8f acc[4] = {};

  const int arow = mtile * 16 + (lane & 15);
  const int kofs = (lane >> 4) ? 8 : 0;        // A frag layout (16x32 f16)
  const _Float16* abase = xh + (size_t)arow * K2 + kofs;

  // per-thread staging coordinates (16B chunks, 2 per thread)
  int sl[2], sG[2], sr[2], sc[2];
#pragma unroll
  for (int c2 = 0; c2 < 2; ++c2) {
    sl[c2] = (threadIdx.x + c2 * 128) * 8;     // half index in Bsh buffer
    sG[c2] = sl[c2] >> 9;
    int rem = sl[c2] & 511;
    sr[c2] = rem >> 4;
    sc[c2] = rem & 15;
  }

  // prologue: stage k0 = 0 into buffer 0
#pragma unroll
  for (int c2 = 0; c2 < 2; ++c2)
    async_stage_b128(Wt + (size_t)sr[c2] * G4 + sG[c2] * HD + nbase + sc[c2],
                     &Bsh[0][sl[c2]]);
  wait_async0();
  __syncthreads();

  int buf = 0;
  for (int k0 = 0; k0 < K2; k0 += 32) {
    // stage next chunk into the other buffer while computing this one
    if (k0 + 32 < K2) {
#pragma unroll
      for (int c2 = 0; c2 < 2; ++c2)
        async_stage_b128(
            Wt + (size_t)(k0 + 32 + sr[c2]) * G4 + sG[c2] * HD + nbase + sc[c2],
            &Bsh[buf ^ 1][sl[c2]]);
    }

    v8h alo = *(const v8h*)(abase + k0);        // K {0..7}/{8..15}
    v8h ahi = *(const v8h*)(abase + k0 + 16);   // K {16..23}/{24..31}
    v16h afrag = cat8(alo, ahi);

#pragma unroll
    for (int G = 0; G < 4; ++G) {
      const _Float16* bp = &Bsh[buf][G * 512 + (lane & 15)];   // lane = N col
      v16h bfrag;
#pragma unroll
      for (int s = 0; s < 16; ++s)
        bfrag[s] = bp[((s < 8) ? (kofs + s) : (16 + kofs + s - 8)) * 16];
      acc[G] = __builtin_amdgcn_wmma_f32_16x16x32_f16(
          false, afrag, false, bfrag, (short)0, acc[G], false, false);
    }

    wait_async0();
    __syncthreads();
    buf ^= 1;
  }

  // Fused LSTM cell (gate order i, f, g, o)
  const int n  = nbase + (lane & 15);
  const float bi = bias[0 * HD + n];
  const float bf = bias[1 * HD + n];
  const float bg = bias[2 * HD + n];
  const float bo = bias[3 * HD + n];
  const int mb = mtile * 16 + ((lane >> 4) << 3);  // C layout: M = r + 8*(lane>=16)
#pragma unroll
  for (int r = 0; r < 8; ++r) {
    const int m = mb + r;
    float iv = sigmf(acc[0][r] + bi);
    float fv = sigmf(acc[1][r] + bf);
    float gv = tanhf(acc[2][r] + bg);
    float ov = sigmf(acc[3][r] + bo);
    const size_t ci = (size_t)m * HD + n;
    float cnew = fv * Cst[ci] + iv * gv;
    Cst[ci] = cnew;
    float h = ov * tanhf(cnew);
    _Float16 hh = (_Float16)h;
    hdstA[(size_t)m * K2 + n] = hh;
    hdstB[(size_t)m * K2 + n] = hh;
    if (feats) feats[(size_t)m * (TT * HD) + n] = hh;
  }
}

// ---------------------------------------------------------------------------
// Head GEMM: Y[65536,256] = feats[65536,512] @ Wh1[512,256] + b1
// block = 256 (8 waves, one M tile each), 64-col N strip (4 tiles/wave).
// Same double-buffered (async) LDS staging for B.
// ---------------------------------------------------------------------------
__global__ __launch_bounds__(256) void head_gemm(
    const _Float16* __restrict__ A,    // [65536,512]
    const _Float16* __restrict__ Bw,   // [512,256]
    const float* __restrict__ b1,
    float* __restrict__ Y)             // [65536,256]
{
  __shared__ __align__(16) _Float16 Bsh[2][32 * 64];   // 2 x 4KB
  const int lane  = threadIdx.x & 31;
  const int w     = threadIdx.x >> 5;
  const int nbase = blockIdx.y * 64;
  const int m0    = blockIdx.x * 128;

  v8f acc[4] = {};
  const int arow = m0 + w * 16 + (lane & 15);
  const int kofs = (lane >> 4) ? 8 : 0;
  const _Float16* abase = A + (size_t)arow * HD + kofs;

  const int sl = threadIdx.x * 8;        // 16B chunk per thread
  const int sr = sl >> 6;
  const int sc = sl & 63;

  async_stage_b128(Bw + (size_t)sr * MIDN + nbase + sc, &Bsh[0][sl]);
  wait_async0();
  __syncthreads();

  int buf = 0;
  for (int k0 = 0; k0 < HD; k0 += 32) {
    if (k0 + 32 < HD)
      async_stage_b128(Bw + (size_t)(k0 + 32 + sr) * MIDN + nbase + sc,
                       &Bsh[buf ^ 1][sl]);

    v16h afrag = cat8(*(const v8h*)(abase + k0), *(const v8h*)(abase + k0 + 16));
#pragma unroll
    for (int j = 0; j < 4; ++j) {
      const _Float16* bp = &Bsh[buf][j * 16 + (lane & 15)];
      v16h bfrag;
#pragma unroll
      for (int s = 0; s < 16; ++s)
        bfrag[s] = bp[((s < 8) ? (kofs + s) : (16 + kofs + s - 8)) * 64];
      acc[j] = __builtin_amdgcn_wmma_f32_16x16x32_f16(
          false, afrag, false, bfrag, (short)0, acc[j], false, false);
    }

    wait_async0();
    __syncthreads();
    buf ^= 1;
  }

  const int mb = m0 + w * 16 + ((lane >> 4) << 3);
#pragma unroll
  for (int j = 0; j < 4; ++j) {
    const int n = nbase + j * 16 + (lane & 15);
    const float bb = b1[n];
#pragma unroll
    for (int r = 0; r < 8; ++r)
      Y[(size_t)(mb + r) * MIDN + n] = acc[j][r] + bb;
  }
}

// ---------------------------------------------------------------------------
// BatchNorm stats: per-column sum / sumsq over 65536 rows (coalesced).
// ---------------------------------------------------------------------------
__global__ __launch_bounds__(256) void bn_stats(
    const float* __restrict__ Y, float* __restrict__ sums, float* __restrict__ sumsq)
{
  const int c  = threadIdx.x;
  const int r0 = blockIdx.x * 256;
  float s = 0.0f, q = 0.0f;
  for (int l = 0; l < 256; ++l) {
    float v = Y[(size_t)(r0 + l) * MIDN + c];
    s += v;
    q += v * v;
  }
  atomicAdd(&sums[c], s);
  atomicAdd(&sumsq[c], q);
}

__global__ __launch_bounds__(256) void bn_coef(
    const float* __restrict__ sums, const float* __restrict__ sumsq,
    const float* __restrict__ gamma, const float* __restrict__ beta,
    float* __restrict__ abn, float* __restrict__ bbn)
{
  const int c = threadIdx.x;
  const float inv_n = 1.0f / (float)NROWS;
  float mu  = sums[c] * inv_n;
  float var = sumsq[c] * inv_n - mu * mu;   // biased var = E[y^2] - mu^2
  float a   = rsqrtf(var + 1e-5f) * gamma[c];
  abn[c] = a;
  bbn[c] = beta[c] - mu * a;
}

// ---------------------------------------------------------------------------
// Final: out[r,o] = relu(norm(Y[r,:])) @ W2.T + b2 ; 32 rows per block via LDS.
// ---------------------------------------------------------------------------
__global__ __launch_bounds__(256) void head_final(
    const float* __restrict__ Y, const float* __restrict__ abn,
    const float* __restrict__ bbn, const float* __restrict__ W2,
    const float* __restrict__ b2, float* __restrict__ out)
{
  __shared__ float Ysh[32 * 257];   // padded to avoid bank conflicts
  __shared__ float W2sh[6 * 257];
  const int t  = threadIdx.x;
  const int r0 = blockIdx.x * 32;
  const float a  = abn[t];
  const float bb = bbn[t];
  for (int l = 0; l < 32; ++l) {
    float v = Y[(size_t)(r0 + l) * MIDN + t] * a + bb;
    Ysh[l * 257 + t] = v > 0.0f ? v : 0.0f;
  }
#pragma unroll
  for (int o = 0; o < 6; ++o) W2sh[o * 257 + t] = W2[o * MIDN + t];
  __syncthreads();
  if (t < 192) {
    int r = t / 6, o = t - 6 * r;
    float s = b2[o];
    for (int c = 0; c < MIDN; ++c) s += Ysh[r * 257 + c] * W2sh[o * 257 + c];
    out[(size_t)(r0 + r) * 6 + o] = s;
  }
}

// ---------------------------------------------------------------------------
extern "C" void kernel_launch(void* const* d_in, const int* in_sizes, int n_in,
                              void* d_out, int out_size, void* d_ws, size_t ws_size,
                              hipStream_t stream)
{
  (void)in_sizes; (void)n_in; (void)out_size; (void)ws_size;
  const float* rt    = (const float*)d_in[0];
  // d_in[1] audio: only defines T, unused by the math
  const float* Wih0  = (const float*)d_in[2];
  const float* Whh0  = (const float*)d_in[3];
  const float* bih0  = (const float*)d_in[4];
  const float* bhh0  = (const float*)d_in[5];
  const float* Wih1  = (const float*)d_in[6];
  const float* Whh1  = (const float*)d_in[7];
  const float* bih1  = (const float*)d_in[8];
  const float* bhh1  = (const float*)d_in[9];
  const float* W1    = (const float*)d_in[10];
  const float* b1    = (const float*)d_in[11];
  const float* gamma = (const float*)d_in[12];
  const float* beta  = (const float*)d_in[13];
  const float* W2    = (const float*)d_in[14];
  const float* b2    = (const float*)d_in[15];
  float* out = (float*)d_out;

  char* ws = (char*)d_ws;
  size_t off = 0;
  auto take = [&](size_t bytes) -> char* {
    char* p = ws + off;
    off += (bytes + 255) & ~(size_t)255;
    return p;
  };
  _Float16* Wt0   = (_Float16*)take((size_t)K2 * G4 * 2);     // 4 MB
  _Float16* Wt1   = (_Float16*)take((size_t)K2 * G4 * 2);     // 4 MB
  _Float16* Wh1   = (_Float16*)take((size_t)HD * MIDN * 2);
  float*    bias0 = (float*)take(G4 * 4);
  float*    bias1 = (float*)take(G4 * 4);
  _Float16* xh0   = (_Float16*)take((size_t)BAT * K2 * 2);
  _Float16* xh1   = (_Float16*)take((size_t)BAT * K2 * 2);
  float*    c0    = (float*)take((size_t)BAT * HD * 4);
  float*    c1    = (float*)take((size_t)BAT * HD * 4);
  float*    sums  = (float*)take(MIDN * 4);
  float*    sumsq = (float*)take(MIDN * 4);
  float*    abn   = (float*)take(MIDN * 4);
  float*    bbn   = (float*)take(MIDN * 4);
  _Float16* feats = (_Float16*)take((size_t)NROWS * HD * 2);  // 64 MB
  float*    Y     = (float*)take((size_t)NROWS * MIDN * 4);   // 64 MB

  prep_weights<<<4096, 256, 0, stream>>>(Wih0, Whh0, Wih1, Whh1, W1, Wt0, Wt1, Wh1);
  init_state<<<512, 256, 0, stream>>>(rt, bih0, bhh0, bih1, bhh1,
                                      xh0, xh1, c0, c1, bias0, bias1, sums, sumsq);

  dim3 gs(HD / 16, BAT / 64);   // 32 n-strips x 2 m-groups
  for (int t = 0; t < TT; ++t) {
    // layer 0: h0 -> layer1 x region, and own h region
    lstm_step<<<gs, 128, 0, stream>>>(xh0, Wt0, bias0, c0, xh1, xh0 + HD, (_Float16*)nullptr);
    // layer 1: h1 -> next-step layer0 x region, own h region, and feats[(b*T+t), :]
    lstm_step<<<gs, 128, 0, stream>>>(xh1, Wt1, bias1, c1, xh0, xh1 + HD,
                                      feats + (size_t)t * HD);
  }

  head_gemm<<<dim3(NROWS / 128, MIDN / 64), 256, 0, stream>>>(feats, Wh1, b1, Y);
  bn_stats<<<NROWS / 256, 256, 0, stream>>>(Y, sums, sumsq);
  bn_coef<<<1, 256, 0, stream>>>(sums, sumsq, gamma, beta, abn, bbn);
  head_final<<<NROWS / 32, 256, 0, stream>>>(Y, abn, bbn, W2, b2, out);
}
